// ExRestSelfAtten_88931592831625
// MI455X (gfx1250) — matrix operand order repaired and checked by reading, not compiled, not run on hardware
//
#include <hip/hip_runtime.h>
#include <math.h>

// ---------------------------------------------------------------------------
// Local-window self-attention block for MI455X (gfx1250, wave32, WMMA).
//   h   = relu(x @ W0 + b0)                 [8192,512]
//   Q/K/V = h @ Wq/Wk/Wv                    [8192,512]   (windows of K/V == shifted rows)
//   dist[t,j] = Q[t]·K[t+16-j]/sqrt(512); attn=softmax; ctx = sum attn*V
//   out = relu(ctx@W1+b1) @ Wout + bout
// Big GEMMs + banded attention run on v_wmma_f32_16x16x32_f16 (f16 operands,
// f32 accumulate). GEMM B-tiles are streamed global->LDS with CDNA5
// GLOBAL_LOAD_ASYNC_TO_LDS_B128 (ASYNCcnt) and double-buffered so the next
// k-slab loads behind the current slab's WMMAs.
// ---------------------------------------------------------------------------

typedef __attribute__((ext_vector_type(16))) _Float16 v16h;
typedef __attribute__((ext_vector_type(8)))  _Float16 v8h;
typedef __attribute__((ext_vector_type(8)))  float    v8f;

#define WMMA_F16(a, b, c) \
  __builtin_amdgcn_wmma_f32_16x16x32_f16(false, (a), false, (b), (short)0, (c), false, false)

// A fragment (16x32 f16): lane L holds row (L&15); K halves {0..15} for
// lanes 0-15 and {16..31} for lanes 16-31. Caller passes p already
// offset by (lane>>4)*16 halves; second chunk sits 8 halves later.
static __device__ __forceinline__ v16h make_afrag(const _Float16* p) {
  v8h lo = *(const v8h*)p;
  v8h hi = *(const v8h*)(p + 8);
  v16h a;
#pragma unroll
  for (int i = 0; i < 8; ++i) { a[i] = lo[i]; a[i + 8] = hi[i]; }
  return a;
}

// Async global->LDS copy of 32B (two B128 transfers), tracked by ASYNCcnt.
static __device__ __forceinline__ void async_stage32(const _Float16* src, unsigned ldsa) {
  asm volatile(
      "global_load_async_to_lds_b128 %0, %2, off\n\t"
      "global_load_async_to_lds_b128 %1, %3, off"
      :
      : "v"(ldsa), "v"(ldsa + 16u), "v"(src), "v"(src + 8)
      : "memory");
}
static __device__ __forceinline__ void wait_async_le2() {
  asm volatile("s_wait_asynccnt 0x2" ::: "memory");
}
static __device__ __forceinline__ void wait_async_0() {
  asm volatile("s_wait_asynccnt 0x0" ::: "memory");
}

// ------------------------------ prep kernels -------------------------------

__global__ __launch_bounds__(256) void cvt_f32_f16(const float* __restrict__ s,
                                                   _Float16* __restrict__ d, int n) {
  int i = blockIdx.x * 256 + threadIdx.x;
  if (i < n) d[i] = (_Float16)s[i];
}

// src [R][C] f32 -> dst [C][R] f16 (LDS-tiled, coalesced both sides)
__global__ __launch_bounds__(256) void tr_f32_f16(const float* __restrict__ src,
                                                  _Float16* __restrict__ dst,
                                                  int R, int C) {
  __shared__ float tile[32][33];
  int tx = threadIdx.x & 31, ty = threadIdx.x >> 5;
  int r0 = blockIdx.y * 32, c0 = blockIdx.x * 32;
#pragma unroll
  for (int i = 0; i < 4; ++i) {
    int r = r0 + ty + 8 * i;
    if (r < R && c0 + tx < C) tile[ty + 8 * i][tx] = src[(size_t)r * C + c0 + tx];
  }
  __syncthreads();
#pragma unroll
  for (int i = 0; i < 4; ++i) {
    int oc = c0 + ty + 8 * i;
    if (oc < C && r0 + tx < R) dst[(size_t)oc * R + r0 + tx] = (_Float16)tile[tx][ty + 8 * i];
  }
}

// src [R][C] f16 -> dst [C][R] f16
__global__ __launch_bounds__(256) void tr_f16(const _Float16* __restrict__ src,
                                              _Float16* __restrict__ dst,
                                              int R, int C) {
  __shared__ _Float16 tile[32][34];
  int tx = threadIdx.x & 31, ty = threadIdx.x >> 5;
  int r0 = blockIdx.y * 32, c0 = blockIdx.x * 32;
#pragma unroll
  for (int i = 0; i < 4; ++i) {
    int r = r0 + ty + 8 * i;
    if (r < R && c0 + tx < C) tile[ty + 8 * i][tx] = src[(size_t)r * C + c0 + tx];
  }
  __syncthreads();
#pragma unroll
  for (int i = 0; i < 4; ++i) {
    int oc = c0 + ty + 8 * i;
    if (oc < C && r0 + tx < R) dst[(size_t)oc * R + r0 + tx] = tile[tx][ty + 8 * i];
  }
}

// ------------------------------ WMMA GEMM ----------------------------------
// C[m,n] = act(sum_k A[m,k] * Bt[n,k] + bias[n]), A:[M][K] f16, Bt:[N][K] f16.
// Block = 128 threads = 4 waves; block tile 64 rows x 64 cols; each wave owns
// 16 rows x 64 cols (4 f32 accumulators). Bt k-slabs streamed to LDS with
// async-to-LDS loads, double-buffered: slab kc+1 loads behind slab kc's WMMAs.
__global__ __launch_bounds__(128) void gemm16(const _Float16* __restrict__ A,
                                              const _Float16* __restrict__ Bt,
                                              const float* __restrict__ bias,
                                              _Float16* __restrict__ C,
                                              int M, int K, int N, int relu) {
  __shared__ _Float16 bt[2][64][32];
  const int lane = threadIdx.x & 31, wid = threadIdx.x >> 5;
  const int g = lane >> 4, lm = lane & 15;
  const int m0 = blockIdx.x * 64 + wid * 16;
  const int n0 = blockIdx.y * 64;

  // per-thread staging slot: 16 halves (32B) of Bt row n0+nl
  const int nl = threadIdx.x >> 1;
  const int ho = (threadIdx.x & 1) * 16;
  const unsigned lds0 = (unsigned)(uintptr_t)&bt[0][nl][ho];
  const unsigned ldsStride = (unsigned)(sizeof(_Float16) * 64 * 32);  // 4KB per buffer
  const _Float16* bsrc = Bt + (size_t)(n0 + nl) * K + ho;
  const _Float16* arow = A + (size_t)(m0 + lm) * K;

  v8f acc[4];
#pragma unroll
  for (int j = 0; j < 4; ++j) acc[j] = (v8f){0.f, 0.f, 0.f, 0.f, 0.f, 0.f, 0.f, 0.f};

  const int kchunks = K >> 5;
  async_stage32(bsrc, lds0);  // prime buffer 0

  for (int kc = 0; kc < kchunks; ++kc) {
    if (kc + 1 < kchunks) {
      async_stage32(bsrc + (kc + 1) * 32, lds0 + ((kc + 1) & 1) * ldsStride);
      wait_async_le2();  // in-order: slab kc is resident, kc+1 still in flight
    } else {
      wait_async_0();
    }
    __syncthreads();

    if (kc + 2 < kchunks)  // prefetch A stream two slabs ahead
      __builtin_prefetch(arow + (kc + 2) * 32, 0, 1);

    v16h a = make_afrag(arow + kc * 32 + g * 16);
#pragma unroll
    for (int j = 0; j < 4; ++j) {
      v16h bf = *(const v16h*)&bt[kc & 1][16 * j + lm][g * 16];
      acc[j] = WMMA_F16(a, bf, acc[j]);
    }
    __syncthreads();  // all waves done reading before this buffer is refilled
  }

#pragma unroll
  for (int j = 0; j < 4; ++j) {
    int n = n0 + 16 * j + lm;
    float bb = bias ? bias[n] : 0.f;
#pragma unroll
    for (int r = 0; r < 8; ++r) {
      int m = m0 + 8 * g + r;
      float v = acc[j][r] + bb;
      if (relu) v = fmaxf(v, 0.f);
      C[(size_t)m * N + n] = (_Float16)v;
    }
  }
}

// --------------------------- fused banded attention ------------------------
// Per 16-row tile: S[16][48] = Q @ K^T over the band (3 WMMA col-tiles x 16 k
// chunks), softmax per row (writes attn output), then ctx = attn_band @ V via
// WMMA with V^T fragments (2 k-chunks x 32 n-tiles).
__global__ __launch_bounds__(128) void attn_fused(const _Float16* __restrict__ Q,
                                                  const _Float16* __restrict__ Km,
                                                  const _Float16* __restrict__ Vt,
                                                  float* __restrict__ attn_out,
                                                  _Float16* __restrict__ ctx) {
  __shared__ float    Sf[4][16][49];
  __shared__ _Float16 Aw[4][16][64];
  const int lane = threadIdx.x & 31, wid = threadIdx.x >> 5;
  const int g = lane >> 4, lm = lane & 15;
  const int tile = blockIdx.x * 4 + wid;          // 0..511
  const int b = tile >> 7;
  const int t0 = (tile & 127) * 16;
  const size_t rowbase = (size_t)b * 2048;

  // ---- phase 1: banded Q.K^T -> 3 accumulator tiles (cols p = t0-16..t0+31)
  v8f s[3];
#pragma unroll
  for (int j = 0; j < 3; ++j) s[j] = (v8f){0.f, 0.f, 0.f, 0.f, 0.f, 0.f, 0.f, 0.f};

  for (int kc = 0; kc < 16; ++kc) {
    const _Float16* qp = Q + (rowbase + t0 + lm) * 512 + kc * 32 + g * 16;
    v16h a = make_afrag(qp);
#pragma unroll
    for (int j = 0; j < 3; ++j) {
      int pl = t0 - 16 + 16 * j + lm;   // window position (per-batch local)
      v16h bf = {};
      if (pl >= 0 && pl < 2048)
        bf = *(const v16h*)(Km + (rowbase + pl) * 512 + kc * 32 + g * 16);
      s[j] = WMMA_F16(a, bf, s[j]);
    }
  }

  // dump S to LDS; zero the band-weight tile (cols 48..63 stay zero = K pad)
#pragma unroll
  for (int j = 0; j < 3; ++j)
#pragma unroll
    for (int r = 0; r < 8; ++r)
      Sf[wid][r + 8 * g][16 * j + lm] = s[j][r];
  {
    _Float16* awf = &Aw[wid][0][0];
    for (int i = lane; i < 1024; i += 32) awf[i] = (_Float16)0.f;
  }
  __syncthreads();

  // ---- phase 2: per-row softmax over the 33-wide band (lanes 0..15 = rows)
  if (lane < 16) {
    const int m = lane;
    const size_t trow = rowbase + t0 + m;
    const float invs = 0.04419417382415922f;     // 1/sqrt(512)
    float mx = -3.4e38f;
    for (int i = 0; i < 33; ++i) mx = fmaxf(mx, Sf[wid][m][m + i] * invs);
    float sum = 0.f;
    for (int i = 0; i < 33; ++i) sum += __expf(Sf[wid][m][m + i] * invs - mx);
    float rinv = 1.f / sum;
    for (int i = 0; i < 33; ++i) {               // col c=m+i is p=t0+c-16 -> jw=32-i
      float w = __expf(Sf[wid][m][m + i] * invs - mx) * rinv;
      attn_out[trow * 33 + (32 - i)] = w;
      Aw[wid][m][m + i] = (_Float16)w;
    }
  }
  __syncthreads();

  // ---- phase 3: ctx = attn_band(16x64) @ V(64x512) using V^T fragments
  v16h aw0 = make_afrag(&Aw[wid][lm][0 * 32 + g * 16]);
  v16h aw1 = make_afrag(&Aw[wid][lm][1 * 32 + g * 16]);
  const int p0 = t0 - 16;
  for (int nt = 0; nt < 32; ++nt) {
    v8f c = (v8f){0.f, 0.f, 0.f, 0.f, 0.f, 0.f, 0.f, 0.f};
#pragma unroll
    for (int kc = 0; kc < 2; ++kc) {
      int pl = p0 + kc * 32 + g * 16;            // 16-aligned run: fully in or out
      v16h bf = {};
      if (pl >= 0 && pl < 2048)
        bf = *(const v16h*)(Vt + (size_t)(nt * 16 + lm) * 8192 + rowbase + pl);
      c = WMMA_F16(kc ? aw1 : aw0, bf, c);
    }
#pragma unroll
    for (int r = 0; r < 8; ++r)
      ctx[(rowbase + t0 + r + 8 * g) * 512 + nt * 16 + lm] = (_Float16)c[r];
  }
}

// ------------------------------- output head -------------------------------
__global__ __launch_bounds__(256) void out_head(const _Float16* __restrict__ h2,
                                                const float* __restrict__ Wout,
                                                const float* __restrict__ bout,
                                                float* __restrict__ out, int Mtot) {
  int idx = blockIdx.x * 256 + threadIdx.x;
  int row = idx >> 1, n = idx & 1;
  if (row >= Mtot) return;
  float acc = bout[n];
  const _Float16* hp = h2 + (size_t)row * 512;
#pragma unroll 4
  for (int k = 0; k < 512; ++k) acc += (float)hp[k] * Wout[k * 2 + n];
  out[row * 2 + n] = acc;
}

// ------------------------------- host driver -------------------------------

extern "C" void kernel_launch(void* const* d_in, const int* in_sizes, int n_in,
                              void* d_out, int out_size, void* d_ws, size_t ws_size,
                              hipStream_t stream) {
  const float* x    = (const float*)d_in[0];
  const float* Wq   = (const float*)d_in[1];
  const float* Wk   = (const float*)d_in[2];
  const float* Wv   = (const float*)d_in[3];
  const float* W0   = (const float*)d_in[4];
  const float* b0   = (const float*)d_in[5];
  const float* W1   = (const float*)d_in[6];
  const float* b1   = (const float*)d_in[7];
  const float* Wout = (const float*)d_in[8];
  const float* bout = (const float*)d_in[9];

  float* out = (float*)d_out;
  float* attn_out = out + (size_t)4 * 2048 * 2;

  const int M = 8192, H = 512, DIN = 256;

  char* ws = (char*)d_ws;
  size_t off = 0;
  auto alloc = [&](size_t bytes) -> char* {
    char* p = ws + off;
    off += (bytes + 255) & ~(size_t)255;
    return p;
  };
  _Float16* x16  = (_Float16*)alloc((size_t)M * DIN * 2);
  _Float16* W0t  = (_Float16*)alloc((size_t)H * DIN * 2);
  _Float16* Wqt  = (_Float16*)alloc((size_t)H * H * 2);
  _Float16* Wkt  = (_Float16*)alloc((size_t)H * H * 2);
  _Float16* Wvt  = (_Float16*)alloc((size_t)H * H * 2);
  _Float16* W1t  = (_Float16*)alloc((size_t)H * H * 2);
  _Float16* h16  = (_Float16*)alloc((size_t)M * H * 2);
  _Float16* Q16  = (_Float16*)alloc((size_t)M * H * 2);
  _Float16* K16  = (_Float16*)alloc((size_t)M * H * 2);
  _Float16* V16  = (_Float16*)alloc((size_t)M * H * 2);
  _Float16* Vt16 = (_Float16*)alloc((size_t)H * M * 2);
  _Float16* ctx16 = (_Float16*)alloc((size_t)M * H * 2);
  _Float16* h216  = (_Float16*)alloc((size_t)M * H * 2);

  // prep: conversions + weight transposes to [N][K] f16
  cvt_f32_f16<<<(M * DIN + 255) / 256, 256, 0, stream>>>(x, x16, M * DIN);
  tr_f32_f16<<<dim3(H / 32, DIN / 32), 256, 0, stream>>>(W0, W0t, DIN, H);
  tr_f32_f16<<<dim3(H / 32, H / 32), 256, 0, stream>>>(Wq, Wqt, H, H);
  tr_f32_f16<<<dim3(H / 32, H / 32), 256, 0, stream>>>(Wk, Wkt, H, H);
  tr_f32_f16<<<dim3(H / 32, H / 32), 256, 0, stream>>>(Wv, Wvt, H, H);
  tr_f32_f16<<<dim3(H / 32, H / 32), 256, 0, stream>>>(W1, W1t, H, H);

  dim3 gGemm(M / 64, H / 64);
  // h = relu(x @ W0 + b0)
  gemm16<<<gGemm, 128, 0, stream>>>(x16, W0t, b0, h16, M, DIN, H, 1);
  // Q/K/V = h @ W{q,k,v}
  gemm16<<<gGemm, 128, 0, stream>>>(h16, Wqt, nullptr, Q16, M, H, H, 0);
  gemm16<<<gGemm, 128, 0, stream>>>(h16, Wkt, nullptr, K16, M, H, H, 0);
  gemm16<<<gGemm, 128, 0, stream>>>(h16, Wvt, nullptr, V16, M, H, H, 0);
  // V^T for the attn.V WMMA B-fragments
  tr_f16<<<dim3(H / 32, M / 32), 256, 0, stream>>>(V16, Vt16, M, H);

  // fused banded attention: attn probs -> d_out tail, ctx -> f16
  attn_fused<<<128, 128, 0, stream>>>(Q16, K16, Vt16, attn_out, ctx16);

  // h2 = relu(ctx @ W1 + b1)
  gemm16<<<gGemm, 128, 0, stream>>>(ctx16, W1t, b1, h216, M, H, H, 1);
  // out = h2 @ Wout + bout  (N=2: VALU head)
  out_head<<<(2 * M + 255) / 256, 256, 0, stream>>>(h216, Wout, bout, out, M);
}
